// AttentionalPropagation_43284680409336
// MI455X (gfx1250) — matrix operand order repaired and checked by reading, not compiled, not run on hardware
//
#include <hip/hip_runtime.h>
#include <hip/hip_bf16.h>
#include <math.h>

typedef __bf16 bf16_t;
typedef __attribute__((ext_vector_type(16))) __bf16 v16bf;
typedef __attribute__((ext_vector_type(8)))  __bf16 v8bf;
typedef __attribute__((ext_vector_type(8)))  float  v8f;

union Frag16 { v16bf v; v8bf h[2]; };

constexpr int BB = 64;     // batch
constexpr int D  = 256;    // model dim
constexpr int NN = 512;    // sequence
constexpr int H  = 4;      // heads
constexpr int DH = 64;     // head dim
constexpr int D2 = 512;    // 2*D

// ---------------- workspace layout (bytes) ----------------
constexpr size_t OFF_XBT   = 0;                                      // (B,N,D) bf16
constexpr size_t OFF_SBT   = OFF_XBT  + (size_t)BB*NN*D*2;           // (B,N,D) bf16
constexpr size_t OFF_WQ    = OFF_SBT  + (size_t)BB*NN*D*2;           // (D,D) bf16
constexpr size_t OFF_WK    = OFF_WQ   + (size_t)D*D*2;
constexpr size_t OFF_WV    = OFF_WK   + (size_t)D*D*2;
constexpr size_t OFF_WM    = OFF_WV   + (size_t)D*D*2;
constexpr size_t OFF_W1    = OFF_WM   + (size_t)D*D*2;               // (2D,2D) bf16
constexpr size_t OFF_W2    = OFF_W1   + (size_t)D2*D2*2;             // (D,2D) bf16
constexpr size_t OFF_QH    = OFF_W2   + (size_t)D*D2*2;              // (B,H,N,dh) bf16
constexpr size_t OFF_KH    = OFF_QH   + (size_t)BB*H*NN*DH*2;        // (B,H,N,dh) bf16
constexpr size_t OFF_VT    = OFF_KH   + (size_t)BB*H*NN*DH*2;        // (B,H,dh,N) bf16
constexpr size_t OFF_DMOD  = OFF_VT   + (size_t)BB*H*NN*DH*2;        // (B,N,N) f32
constexpr size_t OFF_MSGT  = OFF_DMOD + (size_t)BB*NN*NN*4;          // (B,N,D) bf16
constexpr size_t OFF_MSGMT = OFF_MSGT + (size_t)BB*NN*D*2;           // (B,N,D) bf16
constexpr size_t OFF_H1T   = OFF_DMOD;   // reuse: dmod dead after attention; (B,N,2D) f32
constexpr size_t OFF_HT    = OFF_QH;     // reuse: Q/K dead after attention; (B,N,2D) bf16

// ---------------- fragment loaders (layouts per CDNA5 ISA) ----------------
// A 16x32 (MxK): lane m holds K = half*8+0..7 and 16+half*8+0..7
__device__ __forceinline__ v16bf load_frag_a(const bf16_t* arow, int k0, int half) {
  Frag16 a;
  a.h[0] = *(const v8bf*)(arow + k0 + half * 8);
  a.h[1] = *(const v8bf*)(arow + k0 + 16 + half * 8);
  return a.v;
}
// B 32x16 (KxN): lane n holds K = half*8+0..7 and 16+half*8+0..7 (B supplied transposed, K contiguous)
__device__ __forceinline__ v16bf load_frag_b(const bf16_t* brow, int k0, int half) {
  Frag16 b;
  b.h[0] = *(const v8bf*)(brow + k0 + half * 8);
  b.h[1] = *(const v8bf*)(brow + k0 + 16 + half * 8);
  return b.v;
}

// ---------------- wave-level 64x32 GEMM: C = A(64xK) * B(Kx32) ----------------
// A row-major (lda); BT: 32 rows (C columns) x K contiguous (ldbt).
// acc[i][j]: i = 16-row subtile, j = 16-col subtile.
__device__ __forceinline__ void gemm64x32(const bf16_t* A, int lda,
                                          const bf16_t* BT, int ldbt,
                                          int K, int lane, v8f (&acc)[4][2]) {
  const int half = (lane >> 4) & 1, m = lane & 15;
  const bf16_t* brow0 = BT + (size_t)m * ldbt;
  const bf16_t* brow1 = BT + (size_t)(16 + m) * ldbt;
  for (int k0 = 0; k0 < K; k0 += 32) {
    v16bf b0 = load_frag_b(brow0, k0, half);
    v16bf b1 = load_frag_b(brow1, k0, half);
#pragma unroll
    for (int i = 0; i < 4; ++i) {
      const bf16_t* arow = A + (size_t)(i * 16 + m) * lda;
      v16bf a = load_frag_a(arow, k0, half);
      acc[i][0] = __builtin_amdgcn_wmma_f32_16x16x32_bf16(false, a, false, b0,
                                                          (short)0, acc[i][0], false, false);
      acc[i][1] = __builtin_amdgcn_wmma_f32_16x16x32_bf16(false, a, false, b1,
                                                          (short)0, acc[i][1], false, false);
    }
  }
}

// ---------------- prep kernels ----------------
__global__ void k_cvt_bf16(const float* __restrict__ in, bf16_t* __restrict__ out, int count) {
  int i = blockIdx.x * blockDim.x + threadIdx.x;
  if (i < count) out[i] = (bf16_t)in[i];
}

// (B,D,N) f32 -> (B,N,D) bf16
__global__ void k_transpose_bf16(const float* __restrict__ in, bf16_t* __restrict__ out) {
  size_t idx = (size_t)blockIdx.x * blockDim.x + threadIdx.x;
  if (idx >= (size_t)BB * D * NN) return;
  int n = (int)(idx % NN);
  size_t t = idx / NN;
  int c = (int)(t % D);
  int b = (int)(t / D);
  out[((size_t)b * NN + n) * D + c] = (bf16_t)in[idx];
}

// distance + stable-rank (== inverse of stable argsort) + proj_dist scatter
__global__ void k_rankmod(const float* __restrict__ kq, const float* __restrict__ ks,
                          const float* __restrict__ proj, float* __restrict__ dmod) {
  int n = blockIdx.x, b = blockIdx.y;
  __shared__ float dr[NN];
  float qx = kq[((size_t)b * NN + n) * 2 + 0];
  float qy = kq[((size_t)b * NN + n) * 2 + 1];
  for (int j = threadIdx.x; j < NN; j += blockDim.x) {
    float dx = qx - ks[((size_t)b * NN + j) * 2 + 0];
    float dy = qy - ks[((size_t)b * NN + j) * 2 + 1];
    dr[j] = sqrtf(dx * dx + dy * dy);
  }
  __syncthreads();
  for (int j = threadIdx.x; j < NN; j += blockDim.x) {
    float dj = dr[j];
    int rank = 0;
    for (int t = 0; t < NN; ++t) {
      float dt_ = dr[t];
      rank += ((dt_ < dj) || ((dt_ == dj) && (t < j))) ? 1 : 0;
    }
    dmod[((size_t)b * NN + n) * NN + j] = proj[(size_t)n * NN + rank] * dj;
  }
}

// ---------------- Q/K/V projections: one wave per 64x32 output tile ----------------
__global__ void __launch_bounds__(32)
k_qkv(const bf16_t* Wqb, const bf16_t* Wkb, const bf16_t* Wvb,
      const float* bq, const float* bk, const float* bv,
      const bf16_t* xbT, const bf16_t* sbT,
      bf16_t* Qh, bf16_t* Kh, bf16_t* Vt) {
  int nt = blockIdx.x;                 // 16 blocks of 32 columns
  int which = blockIdx.y >> 2;         // 0=q, 1=k, 2=v
  int ot = blockIdx.y & 3;             // 4 blocks of 64 output channels
  int b = blockIdx.z;
  int lane = threadIdx.x & 31;
  const int half = lane >> 4, m = lane & 15;
  const bf16_t* A    = (which == 0 ? Wqb : which == 1 ? Wkb : Wvb) + (size_t)ot * 64 * D;
  const float*  bias = (which == 0 ? bq  : which == 1 ? bk  : bv);
  const bf16_t* BT   = (which == 0 ? xbT : sbT) + ((size_t)b * NN + (size_t)nt * 32) * D;
  v8f acc[4][2] = {};
  gemm64x32(A, D, BT, D, D, lane, acc);
#pragma unroll
  for (int i = 0; i < 4; ++i) {
#pragma unroll
    for (int j = 0; j < 2; ++j) {
      int n = nt * 32 + j * 16 + m;
#pragma unroll
      for (int r = 0; r < 8; ++r) {
        int o = ot * 64 + i * 16 + r + half * 8;   // output channel c = d*H + h
        float val = acc[i][j][r] + bias[o];
        int hh = o & (H - 1), d = o >> 2;
        if (which == 2) {
          Vt[(((size_t)b * H + hh) * DH + d) * NN + n] = (bf16_t)val;   // (B,H,dh,N)
        } else {
          bf16_t* dst = (which == 0 ? Qh : Kh);
          dst[(((size_t)b * H + hh) * NN + n) * DH + d] = (bf16_t)val;  // (B,H,N,dh)
        }
      }
    }
  }
}

// ---------------- fused attention: QK^T * dmod/8 -> softmax -> PV ----------------
__global__ void __launch_bounds__(32)
k_attention(const bf16_t* Qh, const bf16_t* Kh, const bf16_t* Vt,
            const float* dmod, bf16_t* msgT) {
  int nt = blockIdx.x, h = blockIdx.y, b = blockIdx.z;
  int lane = threadIdx.x & 31;
  const int half = lane >> 4, m = lane & 15;
  __shared__ float  S[16][NN];          // 32 KB
  __shared__ bf16_t P[16][NN + 16];     // ~16.5 KB, padded
  __shared__ float  red[32];

  // scores: hoist the Q fragments (K = dh = 64 -> two fragments) out of the key loop
  const bf16_t* qrow = Qh + ((((size_t)b * H + h) * NN + nt * 16) + m) * DH;
  v16bf qa0 = load_frag_a(qrow, 0, half);
  v16bf qa1 = load_frag_a(qrow, 32, half);
  for (int mt = 0; mt < NN / 16; ++mt) {
    const bf16_t* krow = Kh + ((((size_t)b * H + h) * NN + mt * 16) + m) * DH;
    v8f acc = {};
    acc = __builtin_amdgcn_wmma_f32_16x16x32_bf16(false, qa0, false, load_frag_b(krow, 0, half),
                                                  (short)0, acc, false, false);
    acc = __builtin_amdgcn_wmma_f32_16x16x32_bf16(false, qa1, false, load_frag_b(krow, 32, half),
                                                  (short)0, acc, false, false);
    int mk = mt * 16 + m;
#pragma unroll
    for (int r = 0; r < 8; ++r) {
      int nq = nt * 16 + r + half * 8;
      S[r + half * 8][mk] = acc[r] * dmod[((size_t)b * NN + nq) * NN + mk] * 0.125f;
    }
  }
  __syncthreads();

  // softmax: 2 lanes per row, each handles half the columns
  int row = m;
  int c0 = half * (NN / 2);
  float mx = -3.0e38f;
  for (int c = 0; c < NN / 2; ++c) mx = fmaxf(mx, S[row][c0 + c]);
  red[lane] = mx;
  __syncthreads();
  mx = fmaxf(red[row], red[row + 16]);
  __syncthreads();
  float sum = 0.f;
  for (int c = 0; c < NN / 2; ++c) {
    float e = __expf(S[row][c0 + c] - mx);
    S[row][c0 + c] = e;
    sum += e;
  }
  red[lane] = sum;
  __syncthreads();
  float inv = 1.f / (red[row] + red[row + 16]);
  for (int c = 0; c < NN / 2; ++c) P[row][c0 + c] = (bf16_t)(S[row][c0 + c] * inv);
  __syncthreads();

  // PV: O(16 x dh) = P(16 x N) * V(N x dh); each P fragment feeds 4 WMMAs
  const bf16_t* prow = &P[0][0] + (size_t)m * (NN + 16);
  const bf16_t* vrow[4];
#pragma unroll
  for (int dt = 0; dt < 4; ++dt)
    vrow[dt] = Vt + ((((size_t)b * H + h) * DH + dt * 16) + m) * NN;
  v8f acc[4] = {};
  for (int k0 = 0; k0 < NN; k0 += 32) {
    v16bf pfrag = load_frag_a(prow, k0, half);
#pragma unroll
    for (int dt = 0; dt < 4; ++dt) {
      v16bf vfrag = load_frag_b(vrow[dt], k0, half);
      acc[dt] = __builtin_amdgcn_wmma_f32_16x16x32_bf16(false, pfrag, false, vfrag,
                                                        (short)0, acc[dt], false, false);
    }
  }
#pragma unroll
  for (int dt = 0; dt < 4; ++dt) {
    int d = dt * 16 + m;
#pragma unroll
    for (int r = 0; r < 8; ++r) {
      int nq = nt * 16 + r + half * 8;
      msgT[((size_t)b * NN + nq) * D + d * H + h] = (bf16_t)acc[dt][r];  // (B,N,D), c=d*H+h
    }
  }
}

// ---------------- msg projection: msgm = Wm * msg + bm (stored transposed) ----------------
__global__ void __launch_bounds__(32)
k_gemm_m(const bf16_t* Wmb, const float* bm, const bf16_t* msgT, bf16_t* msgmT) {
  int nt = blockIdx.x, ot = blockIdx.y, b = blockIdx.z;
  int lane = threadIdx.x & 31;
  const int half = lane >> 4, m = lane & 15;
  const bf16_t* A  = Wmb + (size_t)ot * 64 * D;
  const bf16_t* BT = msgT + ((size_t)b * NN + (size_t)nt * 32) * D;
  v8f acc[4][2] = {};
  gemm64x32(A, D, BT, D, D, lane, acc);
#pragma unroll
  for (int i = 0; i < 4; ++i)
#pragma unroll
    for (int j = 0; j < 2; ++j) {
      int n = nt * 32 + j * 16 + m;
#pragma unroll
      for (int r = 0; r < 8; ++r) {
        int o = ot * 64 + i * 16 + r + half * 8;
        msgmT[((size_t)b * NN + n) * D + o] = (bf16_t)(acc[i][j][r] + bm[o]);
      }
    }
}

// ---------------- MLP layer 1: h1 = W1 * concat(x, msgm) + b1 (f32, transposed) ----------------
__global__ void __launch_bounds__(32)
k_gemm1(const bf16_t* W1b, const float* b1,
        const bf16_t* xbT, const bf16_t* msgmT, float* h1T) {
  int nt = blockIdx.x, ot = blockIdx.y, b = blockIdx.z;
  int lane = threadIdx.x & 31;
  const int half = lane >> 4, m = lane & 15;
  const bf16_t* xrow0 = xbT   + ((size_t)b * NN + nt * 32 + m) * D;
  const bf16_t* xrow1 = xbT   + ((size_t)b * NN + nt * 32 + 16 + m) * D;
  const bf16_t* mrow0 = msgmT + ((size_t)b * NN + nt * 32 + m) * D;
  const bf16_t* mrow1 = msgmT + ((size_t)b * NN + nt * 32 + 16 + m) * D;
  v8f acc[4][2] = {};
  for (int k0 = 0; k0 < D2; k0 += 32) {
    const bf16_t* br0 = (k0 < D) ? (xrow0 + k0) : (mrow0 + (k0 - D));   // concat split
    const bf16_t* br1 = (k0 < D) ? (xrow1 + k0) : (mrow1 + (k0 - D));
    v16bf b0 = load_frag_b(br0, 0, half);
    v16bf b1 = load_frag_b(br1, 0, half);
    if (k0 + 64 < D2) __builtin_prefetch(br0 + 64, 0, 1);
#pragma unroll
    for (int i = 0; i < 4; ++i) {
      const bf16_t* arow = W1b + ((size_t)ot * 64 + i * 16 + m) * D2;
      v16bf a = load_frag_a(arow, k0, half);
      acc[i][0] = __builtin_amdgcn_wmma_f32_16x16x32_bf16(false, a, false, b0,
                                                          (short)0, acc[i][0], false, false);
      acc[i][1] = __builtin_amdgcn_wmma_f32_16x16x32_bf16(false, a, false, b1,
                                                          (short)0, acc[i][1], false, false);
    }
  }
#pragma unroll
  for (int i = 0; i < 4; ++i)
#pragma unroll
    for (int j = 0; j < 2; ++j) {
      int n = nt * 32 + j * 16 + m;
#pragma unroll
      for (int r = 0; r < 8; ++r) {
        int o = ot * 64 + i * 16 + r + half * 8;
        h1T[((size_t)b * NN + n) * D2 + o] = acc[i][j][r] + b1[o];
      }
    }
}

// ---------------- LayerNorm (ddof=1, eps on std) + ReLU -> bf16 ----------------
__global__ void k_ln(const float* __restrict__ h1T, const float* __restrict__ lna,
                     const float* __restrict__ lnb, bf16_t* __restrict__ hT) {
  int n = blockIdx.x, b = blockIdx.y;
  int t = threadIdx.x;
  const float* row = h1T + ((size_t)b * NN + n) * D2;
  __shared__ float red[256];
  float v0 = row[t], v1 = row[t + 256];
  red[t] = v0 + v1;
  __syncthreads();
  for (int s = 128; s > 0; s >>= 1) { if (t < s) red[t] += red[t + s]; __syncthreads(); }
  float mean = red[0] * (1.f / D2);
  __syncthreads();
  float d0 = v0 - mean, d1 = v1 - mean;
  red[t] = d0 * d0 + d1 * d1;
  __syncthreads();
  for (int s = 128; s > 0; s >>= 1) { if (t < s) red[t] += red[t + s]; __syncthreads(); }
  float inv = 1.f / (sqrtf(red[0] * (1.f / (D2 - 1))) + 1e-6f);
  bf16_t* orow = hT + ((size_t)b * NN + n) * D2;
  float y0 = fmaxf(lna[t] * d0 * inv + lnb[t], 0.f);
  float y1 = fmaxf(lna[t + 256] * d1 * inv + lnb[t + 256], 0.f);
  orow[t] = (bf16_t)y0;
  orow[t + 256] = (bf16_t)y1;
}

// ---------------- MLP layer 2: out = W2 * h + b2 (coalesced f32 stores) ----------------
__global__ void __launch_bounds__(32)
k_gemm2(const bf16_t* W2b, const float* b2, const bf16_t* hT, float* out) {
  int nt = blockIdx.x, ot = blockIdx.y, b = blockIdx.z;
  int lane = threadIdx.x & 31;
  const int half = lane >> 4, m = lane & 15;
  const bf16_t* A  = W2b + (size_t)ot * 64 * D2;
  const bf16_t* BT = hT + ((size_t)b * NN + (size_t)nt * 32) * D2;
  v8f acc[4][2] = {};
  gemm64x32(A, D2, BT, D2, D2, lane, acc);
#pragma unroll
  for (int i = 0; i < 4; ++i)
#pragma unroll
    for (int j = 0; j < 2; ++j) {
      int n = nt * 32 + j * 16 + m;
#pragma unroll
      for (int r = 0; r < 8; ++r) {
        int o = ot * 64 + i * 16 + r + half * 8;
        out[((size_t)b * D + o) * NN + n] = acc[i][j][r] + b2[o];
      }
    }
}

extern "C" void kernel_launch(void* const* d_in, const int* in_sizes, int n_in,
                              void* d_out, int out_size, void* d_ws, size_t ws_size,
                              hipStream_t stream) {
  const float* x    = (const float*)d_in[0];
  const float* src  = (const float*)d_in[1];
  const float* kq   = (const float*)d_in[2];
  const float* ks   = (const float*)d_in[3];
  const float* Wq   = (const float*)d_in[4];  const float* bq = (const float*)d_in[5];
  const float* Wk   = (const float*)d_in[6];  const float* bk = (const float*)d_in[7];
  const float* Wv   = (const float*)d_in[8];  const float* bv = (const float*)d_in[9];
  const float* Wm   = (const float*)d_in[10]; const float* bm = (const float*)d_in[11];
  const float* proj = (const float*)d_in[12];
  const float* W1   = (const float*)d_in[13]; const float* b1 = (const float*)d_in[14];
  const float* lna  = (const float*)d_in[15]; const float* lnb = (const float*)d_in[16];
  const float* W2   = (const float*)d_in[17]; const float* b2 = (const float*)d_in[18];
  float* out = (float*)d_out;

  char* ws = (char*)d_ws;
  bf16_t* xbT   = (bf16_t*)(ws + OFF_XBT);
  bf16_t* sbT   = (bf16_t*)(ws + OFF_SBT);
  bf16_t* Wqb   = (bf16_t*)(ws + OFF_WQ);
  bf16_t* Wkb   = (bf16_t*)(ws + OFF_WK);
  bf16_t* Wvb   = (bf16_t*)(ws + OFF_WV);
  bf16_t* Wmb   = (bf16_t*)(ws + OFF_WM);
  bf16_t* W1b   = (bf16_t*)(ws + OFF_W1);
  bf16_t* W2b   = (bf16_t*)(ws + OFF_W2);
  bf16_t* Qh    = (bf16_t*)(ws + OFF_QH);
  bf16_t* Kh    = (bf16_t*)(ws + OFF_KH);
  bf16_t* Vt    = (bf16_t*)(ws + OFF_VT);
  float*  dmod  = (float*) (ws + OFF_DMOD);
  bf16_t* msgT  = (bf16_t*)(ws + OFF_MSGT);
  bf16_t* msgmT = (bf16_t*)(ws + OFF_MSGMT);
  float*  h1T   = (float*) (ws + OFF_H1T);
  bf16_t* hT    = (bf16_t*)(ws + OFF_HT);

  // stage 0: precision conversion + transposes
  k_transpose_bf16<<<dim3((BB * D * NN) / 256), 256, 0, stream>>>(x, xbT);
  k_transpose_bf16<<<dim3((BB * D * NN) / 256), 256, 0, stream>>>(src, sbT);
  k_cvt_bf16<<<dim3((D * D) / 256), 256, 0, stream>>>(Wq, Wqb, D * D);
  k_cvt_bf16<<<dim3((D * D) / 256), 256, 0, stream>>>(Wk, Wkb, D * D);
  k_cvt_bf16<<<dim3((D * D) / 256), 256, 0, stream>>>(Wv, Wvb, D * D);
  k_cvt_bf16<<<dim3((D * D) / 256), 256, 0, stream>>>(Wm, Wmb, D * D);
  k_cvt_bf16<<<dim3((D2 * D2) / 256), 256, 0, stream>>>(W1, W1b, D2 * D2);
  k_cvt_bf16<<<dim3((D * D2) / 256), 256, 0, stream>>>(W2, W2b, D * D2);

  // stage 1: distance modulation (argsort-equivalent stable ranks)
  k_rankmod<<<dim3(NN, BB), 256, 0, stream>>>(kq, ks, proj, dmod);

  // stage 2: projections (WMMA, 64x32 tiles)
  k_qkv<<<dim3(NN / 32, 3 * 4, BB), 32, 0, stream>>>(Wqb, Wkb, Wvb, bq, bk, bv,
                                                     xbT, sbT, Qh, Kh, Vt);

  // stage 3: fused attention (WMMA + LDS softmax)
  k_attention<<<dim3(NN / 16, H, BB), 32, 0, stream>>>(Qh, Kh, Vt, dmod, msgT);

  // stage 4: message projection (WMMA)
  k_gemm_m<<<dim3(NN / 32, D / 64, BB), 32, 0, stream>>>(Wmb, bm, msgT, msgmT);

  // stage 5: MLP layer 1 on concat(x, msgm) (WMMA)
  k_gemm1<<<dim3(NN / 32, D2 / 64, BB), 32, 0, stream>>>(W1b, b1, xbT, msgmT, h1T);

  // stage 6: LayerNorm + ReLU
  k_ln<<<dim3(NN, BB), 256, 0, stream>>>(h1T, lna, lnb, hT);

  // stage 7: MLP layer 2 (WMMA) -> output
  k_gemm2<<<dim3(NN / 32, D / 64, BB), 32, 0, stream>>>(W2b, b2, hT, out);
}
